// XLAttention_7438883357277
// MI455X (gfx1250) — matrix operand order repaired and checked by reading, not compiled, not run on hardware
//
#include <hip/hip_runtime.h>

typedef __attribute__((ext_vector_type(16))) __bf16 v16bf;
typedef __attribute__((ext_vector_type(8)))  __bf16 v8bf;
typedef __attribute__((ext_vector_type(8)))  float  v8f;

#define D_MODEL 1024
#define N_HEAD  16
#define D_HEAD  64
#define SEQ_L   2048
#define SEQ_T   4096   // M + L
#define NB      2

// ---------- helpers ----------

static __device__ __forceinline__ __bf16 f2bf(float f) {
  return (__bf16)f;   // native cvt on gfx1250
}

static __device__ __forceinline__ v8f wmma_bf16(v16bf a, v16bf b, v8f c) {
  // D = A(16x32 bf16) * B(32x16 bf16) + C(16x16 f32)
  return __builtin_amdgcn_wmma_f32_16x16x32_bf16(false, a, false, b, (short)0, c,
                                                 false, false);
}

// A fragment (16x32, bf16). Contiguous layout: lanes 0-15 hold K=0..15,
// lanes 16-31 hold K=16..31. Two contiguous 16B chunks per lane,
// placed directly into the fragment registers via a union (no repacking).
static __device__ __forceinline__ v16bf afrag_bf(const __bf16* row, int half) {
  union { v16bf v; v8bf h[2]; } u;
  u.h[0] = *(const v8bf*)(row + half * 16);
  u.h[1] = *(const v8bf*)(row + half * 16 + 8);
  return u.v;
}

// ---------- elementwise fp32 -> bf16 pre-convert (8 elems/thread) ----------

__global__ __launch_bounds__(256) void xlattn_cvt_kernel(
    const float* __restrict__ in, __bf16* __restrict__ out, int n8)
{
  const int i = blockIdx.x * 256 + threadIdx.x;
  if (i >= n8) return;
  const size_t base = (size_t)i * 8;
  v8bf o;
  #pragma unroll
  for (int e = 0; e < 8; ++e) o[e] = f2bf(in[base + e]);
  *(v8bf*)(out + base) = o;
}

// ---------- projection GEMM: Out = X @ W^T (all-bf16 operands) ----------
// 32x64 output tile per wave. mode 0: row-major bf16 out scaled by 1/8 (Q).
// mode 1: row-major bf16 out (K). mode 2: scatter into transposed V layout
// Vt[(b*16+h)*64+d][T].

__global__ __launch_bounds__(128) void xlattn_proj_kernel(
    const __bf16* __restrict__ X, const __bf16* __restrict__ W,
    float scale, int mode, __bf16* __restrict__ Out)
{
  const int wave = threadIdx.x >> 5;
  const int lane = threadIdx.x & 31;
  const int half = lane >> 4;
  const int l15  = lane & 15;
  const int wid  = blockIdx.x * 4 + wave;
  const int r0   = (wid >> 4) * 32;        // 32-row tile
  const int n0   = (wid & 15) * 64;        // 64-col tile

  const __bf16* arow0 = X + (size_t)(r0 + l15) * D_MODEL;
  const __bf16* arow1 = X + (size_t)(r0 + 16 + l15) * D_MODEL;

  v8f c[2][4];
  #pragma unroll
  for (int gg = 0; gg < 2; ++gg)
    #pragma unroll
    for (int f = 0; f < 4; ++f) c[gg][f] = (v8f){};

  for (int k0 = 0; k0 < D_MODEL; k0 += 32) {
    const v16bf a0 = afrag_bf(arow0 + k0, half);
    const v16bf a1 = afrag_bf(arow1 + k0, half);
    const __bf16* wb = W + (size_t)(n0 + l15) * D_MODEL + k0 + half * 16;
    #pragma unroll
    for (int f = 0; f < 4; ++f) {
      const v16bf bfr = *(const v16bf*)(wb + (size_t)f * 16 * D_MODEL);
      c[0][f] = wmma_bf16(a0, bfr, c[0][f]);
      c[1][f] = wmma_bf16(a1, bfr, c[1][f]);
    }
  }

  if (mode != 2) {
    #pragma unroll
    for (int gg = 0; gg < 2; ++gg)
      #pragma unroll
      for (int g = 0; g < 8; ++g) {
        const size_t base =
            (size_t)(r0 + gg * 16 + half * 8 + g) * D_MODEL + n0 + l15;
        #pragma unroll
        for (int f = 0; f < 4; ++f)
          Out[base + f * 16] = f2bf(c[gg][f][g] * scale);
      }
  } else {
    #pragma unroll
    for (int gg = 0; gg < 2; ++gg)
      #pragma unroll
      for (int g = 0; g < 8; ++g) {
        const int rr = r0 + gg * 16 + half * 8 + g;   // flat c-row = j*NB + b
        const int j  = rr >> 1;
        const int bb = rr & 1;
        #pragma unroll
        for (int f = 0; f < 4; ++f) {
          const int col = n0 + f * 16 + l15;          // h*64 + d
          const int hh = col >> 6, d = col & 63;
          Out[(((size_t)bb * N_HEAD + hh) * D_HEAD + d) * SEQ_T + j] =
              f2bf(c[gg][f][g]);
        }
      }
  }
}

// ---------- fused flash attention ----------
// One wave per 16-query tile of one (b,h). Qp pre-scaled by 1/sqrt(Dh).

__global__ __launch_bounds__(128) void xlattn_attn_kernel(
    const __bf16* __restrict__ Qp,   // [L*B, 1024]
    const __bf16* __restrict__ Kp,   // [T*B, 1024]
    const __bf16* __restrict__ Vt,   // [B*H][64][T]
    __bf16* __restrict__ AV)         // [L*B, 1024]
{
  __shared__ __bf16 Plds[4][16][32];
  const int wave = threadIdx.x >> 5;
  const int lane = threadIdx.x & 31;
  const int half = lane >> 4;
  const int l15  = lane & 15;

  const int bh = blockIdx.x >> 5;                    // 32 blocks per (b,h)
  const int it = ((blockIdx.x & 31) << 2) + wave;    // i-tile index
  const int b  = bh >> 4;
  const int h  = bh & 15;
  const int i0 = it * 16;

  const __bf16* qrow = Qp + ((size_t)(i0 + l15) * NB + b) * D_MODEL + h * D_HEAD;
  const v16bf aq0 = afrag_bf(qrow + 0,  half);       // dh 0..31
  const v16bf aq1 = afrag_bf(qrow + 32, half);       // dh 32..63

  v8f o0 = {}, o1 = {}, o2 = {}, o3 = {};
  float mrow[8], lrow[8];
  #pragma unroll
  for (int g = 0; g < 8; ++g) { mrow[g] = -1e30f; lrow[g] = 0.0f; }

  const __bf16* vbase = Vt + (size_t)bh * D_HEAD * SEQ_T;

  for (int j0 = 0; j0 < SEQ_T; j0 += 32) {
    // --- S = Q K^T for two 16-key tiles ---
    const __bf16* krow0 =
        Kp + ((size_t)(j0 + l15) * NB + b) * D_MODEL + h * D_HEAD + half * 16;
    const __bf16* krow1 = krow0 + (size_t)16 * NB * D_MODEL;
    v8f s0 = {}, s1 = {};
    s0 = wmma_bf16(aq0, *(const v16bf*)(krow0),      s0);
    s0 = wmma_bf16(aq1, *(const v16bf*)(krow0 + 32), s0);
    s1 = wmma_bf16(aq0, *(const v16bf*)(krow1),      s1);
    s1 = wmma_bf16(aq1, *(const v16bf*)(krow1 + 32), s1);

    // --- online softmax: C-layout row = g + 8*half, columns across 16 lanes ---
    float p0[8], p1[8], scal[8];
    #pragma unroll
    for (int g = 0; g < 8; ++g) {
      float v = fmaxf(s0[g], s1[g]);
      v = fmaxf(v, __shfl_xor(v, 1, 32));
      v = fmaxf(v, __shfl_xor(v, 2, 32));
      v = fmaxf(v, __shfl_xor(v, 4, 32));
      v = fmaxf(v, __shfl_xor(v, 8, 32));
      const float mnew = fmaxf(mrow[g], v);
      scal[g] = __expf(mrow[g] - mnew);
      mrow[g] = mnew;
      p0[g] = __expf(s0[g] - mnew);
      p1[g] = __expf(s1[g] - mnew);
      float rs = p0[g] + p1[g];
      rs += __shfl_xor(rs, 1, 32);
      rs += __shfl_xor(rs, 2, 32);
      rs += __shfl_xor(rs, 4, 32);
      rs += __shfl_xor(rs, 8, 32);
      lrow[g] = lrow[g] * scal[g] + rs;
    }

    // --- rescale O, stage P through LDS (C-layout -> A-layout transpose) ---
    #pragma unroll
    for (int g = 0; g < 8; ++g) {
      o0[g] *= scal[g]; o1[g] *= scal[g]; o2[g] *= scal[g]; o3[g] *= scal[g];
      Plds[wave][half * 8 + g][l15]      = f2bf(p0[g]);
      Plds[wave][half * 8 + g][l15 + 16] = f2bf(p1[g]);
    }
    asm volatile("s_wait_dscnt 0" ::: "memory");   // within-wave DS in-order
    const v16bf pa = afrag_bf(&Plds[wave][l15][0], half);

    // --- O += P V : V^T rows give contiguous B-fragment loads ---
    const __bf16* vj = vbase + j0 + half * 16;
    o0 = wmma_bf16(pa, *(const v16bf*)(vj + (size_t)(l15 +  0) * SEQ_T), o0);
    o1 = wmma_bf16(pa, *(const v16bf*)(vj + (size_t)(l15 + 16) * SEQ_T), o1);
    o2 = wmma_bf16(pa, *(const v16bf*)(vj + (size_t)(l15 + 32) * SEQ_T), o2);
    o3 = wmma_bf16(pa, *(const v16bf*)(vj + (size_t)(l15 + 48) * SEQ_T), o3);
  }

  #pragma unroll
  for (int g = 0; g < 8; ++g) {
    const float inv = 1.0f / lrow[g];
    const size_t base =
        ((size_t)(i0 + half * 8 + g) * NB + b) * D_MODEL + h * D_HEAD + l15;
    AV[base + 0]  = f2bf(o0[g] * inv);
    AV[base + 16] = f2bf(o1[g] * inv);
    AV[base + 32] = f2bf(o2[g] * inv);
    AV[base + 48] = f2bf(o3[g] * inv);
  }
}

// ---------- output projection: out(f32) = AV(bf16) @ Wc^T(bf16) ----------
// 32x64 tile per wave.

__global__ __launch_bounds__(128) void xlattn_outproj_kernel(
    const __bf16* __restrict__ AV, const __bf16* __restrict__ W,
    float* __restrict__ Out)
{
  const int wave = threadIdx.x >> 5;
  const int lane = threadIdx.x & 31;
  const int half = lane >> 4;
  const int l15  = lane & 15;
  const int wid  = blockIdx.x * 4 + wave;
  const int r0   = (wid >> 4) * 32;
  const int n0   = (wid & 15) * 64;

  const __bf16* arow0 = AV + (size_t)(r0 + l15) * D_MODEL;
  const __bf16* arow1 = AV + (size_t)(r0 + 16 + l15) * D_MODEL;

  v8f c[2][4];
  #pragma unroll
  for (int gg = 0; gg < 2; ++gg)
    #pragma unroll
    for (int f = 0; f < 4; ++f) c[gg][f] = (v8f){};

  for (int k0 = 0; k0 < D_MODEL; k0 += 32) {
    const v16bf a0 = afrag_bf(arow0 + k0, half);
    const v16bf a1 = afrag_bf(arow1 + k0, half);
    const __bf16* wb = W + (size_t)(n0 + l15) * D_MODEL + k0 + half * 16;
    #pragma unroll
    for (int f = 0; f < 4; ++f) {
      const v16bf bfr = *(const v16bf*)(wb + (size_t)f * 16 * D_MODEL);
      c[0][f] = wmma_bf16(a0, bfr, c[0][f]);
      c[1][f] = wmma_bf16(a1, bfr, c[1][f]);
    }
  }

  #pragma unroll
  for (int gg = 0; gg < 2; ++gg)
    #pragma unroll
    for (int g = 0; g < 8; ++g) {
      const size_t base =
          (size_t)(r0 + gg * 16 + half * 8 + g) * D_MODEL + n0 + l15;
      #pragma unroll
      for (int f = 0; f < 4; ++f) Out[base + f * 16] = c[gg][f][g];
    }
}

// ---------- launch ----------

extern "C" void kernel_launch(void* const* d_in, const int* in_sizes, int n_in,
                              void* d_out, int out_size, void* d_ws, size_t ws_size,
                              hipStream_t stream) {
  const float* q    = (const float*)d_in[0];
  const float* kv   = (const float*)d_in[1];
  const float* mems = (const float*)d_in[2];
  const float* w_q  = (const float*)d_in[3];
  const float* w_k  = (const float*)d_in[4];
  const float* w_v  = (const float*)d_in[5];
  const float* w_c  = (const float*)d_in[6];
  float* out = (float*)d_out;

  // ---- workspace layout (bf16 elements) ----
  const size_t nQrows = (size_t)SEQ_L * NB;        // 4096
  const size_t nCrows = (size_t)SEQ_T * NB;        // 8192
  __bf16* Xq  = (__bf16*)d_ws;                     // [4096,1024]   8MB
  __bf16* Xc  = Xq + nQrows * D_MODEL;             // [8192,1024]  16MB
  __bf16* Wqb = Xc + nCrows * D_MODEL;             // [1024,1024]   2MB
  __bf16* Wkb = Wqb + (size_t)D_MODEL * D_MODEL;
  __bf16* Wvb = Wkb + (size_t)D_MODEL * D_MODEL;
  __bf16* Wcb = Wvb + (size_t)D_MODEL * D_MODEL;
  __bf16* Qp  = Wcb + (size_t)D_MODEL * D_MODEL;   // [4096,1024]   8MB
  __bf16* Kp  = Qp + nQrows * D_MODEL;             // [8192,1024]  16MB
  __bf16* Vt  = Kp + nCrows * D_MODEL;             // [B*H][64][T] 16MB
  __bf16* AV  = Vt + (size_t)NB * N_HEAD * D_HEAD * SEQ_T;  // [4096,1024] 8MB

  const dim3 blk128(128), blk256(256);
  const int nq8 = (int)(nQrows * D_MODEL / 8);     // 524288
  const int nw8 = D_MODEL * D_MODEL / 8;           // 131072

  // pre-convert activations and weights to bf16 (Xc = [mems; kv] row concat)
  xlattn_cvt_kernel<<<dim3(nq8 / 256), blk256, 0, stream>>>(q, Xq, nq8);
  xlattn_cvt_kernel<<<dim3(nq8 / 256), blk256, 0, stream>>>(mems, Xc, nq8);
  xlattn_cvt_kernel<<<dim3(nq8 / 256), blk256, 0, stream>>>(kv, Xc + nQrows * D_MODEL, nq8);
  xlattn_cvt_kernel<<<dim3(nw8 / 256), blk256, 0, stream>>>(w_q, Wqb, nw8);
  xlattn_cvt_kernel<<<dim3(nw8 / 256), blk256, 0, stream>>>(w_k, Wkb, nw8);
  xlattn_cvt_kernel<<<dim3(nw8 / 256), blk256, 0, stream>>>(w_v, Wvb, nw8);
  xlattn_cvt_kernel<<<dim3(nw8 / 256), blk256, 0, stream>>>(w_c, Wcb, nw8);

  // projections (Q scaled by 1/sqrt(64) = 0.125)
  xlattn_proj_kernel<<<dim3((int)(nQrows / 32) * (D_MODEL / 64) / 4), blk128, 0, stream>>>(
      Xq, Wqb, 0.125f, 0, Qp);
  xlattn_proj_kernel<<<dim3((int)(nCrows / 32) * (D_MODEL / 64) / 4), blk128, 0, stream>>>(
      Xc, Wkb, 1.0f, 1, Kp);
  xlattn_proj_kernel<<<dim3((int)(nCrows / 32) * (D_MODEL / 64) / 4), blk128, 0, stream>>>(
      Xc, Wvb, 1.0f, 2, Vt);

  // fused flash attention: 32 (b,h) pairs * 32 blocks
  xlattn_attn_kernel<<<dim3(NB * N_HEAD * (SEQ_L / 16) / 4), blk128, 0, stream>>>(
      Qp, Kp, Vt, AV);

  // output projection -> fp32 d_out
  xlattn_outproj_kernel<<<dim3((int)(nQrows / 32) * (D_MODEL / 64) / 4), blk128, 0, stream>>>(
      AV, Wcb, out);
}
